// EfficientVisionAttention_57672820850761
// MI455X (gfx1250) — compile-verified, matrix operand order checked by reading, not run on hardware
//
#include <hip/hip_runtime.h>

// ---------------------------------------------------------------------------
// EfficientVisionAttention for MI455X (gfx1250, wave32, WMMA)
// v2.1: one-time fp32->bf16 conversion (+weight transpose) so all GEMM tiles
// are pure byte copies; async global->LDS staging (ASYNCcnt path); bf16
// operands make X+W L2-resident (78MB+78MB < 192MB L2) and halve tile BW.
// ---------------------------------------------------------------------------

typedef __attribute__((ext_vector_type(16))) __bf16 v16bf;
typedef __attribute__((ext_vector_type(8)))  float  v8f;
typedef int vi4 __attribute__((__vector_size__(16)));   // matches builtin proto

#define DIM        6272          // B*N rows == C cols == K
#define BM         128           // workgroup tile M
#define BN         128           // workgroup tile N
#define BK         32            // K step (matches 16x16x32 bf16 WMMA)
#define KTILES     (DIM / BK)    // 196
#define LDS_STRIDE 40            // bf16 per LDS row (32 + 8 pad), 80B, 16B-aligned

union FragU  { v16bf v; uint4 q[2]; };
union Pack2  { __bf16 h[2]; unsigned int u; };

#if defined(__has_builtin)
#if __has_builtin(__builtin_amdgcn_global_load_async_to_lds_b128) && \
    __has_builtin(__builtin_amdgcn_s_wait_asynccnt)
#define USE_ASYNC_LDS 1
#endif
#endif
#ifndef USE_ASYNC_LDS
#define USE_ASYNC_LDS 0
#endif

__device__ __forceinline__ void copy16_g2l(const void* g, void* l)
{
#if USE_ASYNC_LDS
    __builtin_amdgcn_global_load_async_to_lds_b128(
        (__attribute__((address_space(1))) vi4*)g,
        (__attribute__((address_space(3))) vi4*)l, 0, 0);
#else
    *(uint4*)l = *(const uint4*)g;
#endif
}

__device__ __forceinline__ void wait_g2l()
{
#if USE_ASYNC_LDS
    __builtin_amdgcn_s_wait_asynccnt(0);
#endif
}

// MODE: 0 = plain store, 1 = relu, 2 = relu * Mul[idx] (in-place ok), 3 = + Bias[col]
// A: bf16 row-major [M][K].  Bt: bf16 N-major [N][K] (pre-transposed weights).
template <int MODE>
__global__ __launch_bounds__(256)
void gemm_bf16_wmma(const __bf16* __restrict__ A, const __bf16* __restrict__ Bt,
                    float* __restrict__ Out, const float* __restrict__ Mul,
                    const float* __restrict__ Bias)
{
    __shared__ __bf16 sA[2][BM][LDS_STRIDE];
    __shared__ __bf16 sB[2][BN][LDS_STRIDE];   // [n][k], same layout as sA

    const int tid    = threadIdx.x;
    const int lane   = tid & 31;
    const int wave   = tid >> 5;          // 0..7
    const int waveM  = wave >> 2;         // 0..1  (64-row strip)
    const int waveN  = wave & 3;          // 0..3  (32-col strip)
    const int laneLo = lane & 15;
    const int laneHi = lane >> 4;         // 0/1 selects K-half per ISA A layout

    const int bm = blockIdx.x * BM;
    const int bn = blockIdx.y * BN;

    // staging: each thread copies 32B of A tile and 32B of B tile (2x b128 each)
    const int crow = tid >> 1;            // 0..127
    const int ccol = (tid & 1) * 16;      // element offset 0 or 16 (32B granules)

    const __bf16* gA = A  + (size_t)(bm + crow) * DIM + ccol;
    const __bf16* gB = Bt + (size_t)(bn + crow) * DIM + ccol;

    auto stage = [&](int kt, int buf) {
        const __bf16* ga = gA + kt * BK;
        const __bf16* gb = gB + kt * BK;
        __bf16* la = &sA[buf][crow][ccol];
        __bf16* lb = &sB[buf][crow][ccol];
        copy16_g2l(ga,     la);
        copy16_g2l(ga + 8, la + 8);
        copy16_g2l(gb,     lb);
        copy16_g2l(gb + 8, lb + 8);
        if (kt + 2 < KTILES) {   // global_prefetch_b8, two K-tiles ahead
            __builtin_prefetch(ga + 2 * BK, 0, 1);
            __builtin_prefetch(gb + 2 * BK, 0, 1);
        }
    };

    v8f acc[8] = {};   // 4 (M) x 2 (N) 16x16 f32 accumulators -> 64x32 wave tile

    stage(0, 0);
    wait_g2l();
    __syncthreads();

    for (int kt = 0; kt < KTILES; ++kt) {
        const int buf = kt & 1;
        if (kt + 1 < KTILES) stage(kt + 1, buf ^ 1);

        FragU af[4];
#pragma unroll
        for (int mi = 0; mi < 4; ++mi) {
            const __bf16* base = &sA[buf][waveM * 64 + mi * 16 + laneLo][laneHi * 8];
            af[mi].q[0] = *(const uint4*)base;
            af[mi].q[1] = *(const uint4*)(base + 16);
        }
#pragma unroll
        for (int ni = 0; ni < 2; ++ni) {
            FragU bfrag;
            const __bf16* base = &sB[buf][waveN * 32 + ni * 16 + laneLo][laneHi * 8];
            bfrag.q[0] = *(const uint4*)base;
            bfrag.q[1] = *(const uint4*)(base + 16);
#pragma unroll
            for (int mi = 0; mi < 4; ++mi) {
                acc[mi * 2 + ni] = __builtin_amdgcn_wmma_f32_16x16x32_bf16(
                    /*neg_a=*/false, af[mi].v, /*neg_b=*/false, bfrag.v,
                    /*c_mod=*/(short)0, acc[mi * 2 + ni],
                    /*reuse_a=*/false, /*reuse_b=*/false);
            }
        }
        wait_g2l();          // next tile's async copies done (overlapped with WMMA)
        __syncthreads();
    }

    // ---- epilogue: C/D layout -> row = base + vgpr + 8*laneHi, col = laneLo ----
#pragma unroll
    for (int mi = 0; mi < 4; ++mi) {
#pragma unroll
        for (int ni = 0; ni < 2; ++ni) {
            v8f a = acc[mi * 2 + ni];
            const int gcol  = bn + waveN * 32 + ni * 16 + laneLo;
            const int grow0 = bm + waveM * 64 + mi * 16 + laneHi * 8;
#pragma unroll
            for (int r = 0; r < 8; ++r) {
                size_t idx = (size_t)(grow0 + r) * DIM + gcol;
                float v = a[r];
                if (MODE == 1) v = fmaxf(v, 0.0f);
                if (MODE == 2) v = fmaxf(v, 0.0f) * Mul[idx];
                if (MODE == 3) v = v + Bias[gcol];
                Out[idx] = v;
            }
        }
    }
}

// fp32 -> bf16, streaming (for X)
__global__ __launch_bounds__(256)
void convert_bf16(const float* __restrict__ in, __bf16* __restrict__ out)
{
    const size_t i = ((size_t)blockIdx.x * 256 + threadIdx.x) * 4;
    float4 f = *(const float4*)(in + i);
    Pack2 a, b;
    a.h[0] = (__bf16)f.x; a.h[1] = (__bf16)f.y;
    b.h[0] = (__bf16)f.z; b.h[1] = (__bf16)f.w;
    *(uint2*)(out + i) = make_uint2(a.u, b.u);
}

// fp32 W[k][n] -> bf16 Wt[n][k], 64x64 LDS tiles (coalesced both sides)
__global__ __launch_bounds__(256)
void convert_tr_bf16(const float* __restrict__ W, __bf16* __restrict__ Wt)
{
    __shared__ __bf16 t[64][72];   // +8 pad
    const int k0 = blockIdx.x * 64;
    const int n0 = blockIdx.y * 64;
    const int r  = threadIdx.x >> 2;          // 0..63
    const int c  = (threadIdx.x & 3) * 16;    // 0,16,32,48

    const float* src = W + (size_t)(k0 + r) * DIM + n0 + c;
#pragma unroll
    for (int i = 0; i < 4; ++i) {
        float4 f = *(const float4*)(src + i * 4);
        t[r][c + i * 4 + 0] = (__bf16)f.x;
        t[r][c + i * 4 + 1] = (__bf16)f.y;
        t[r][c + i * 4 + 2] = (__bf16)f.z;
        t[r][c + i * 4 + 3] = (__bf16)f.w;
    }
    __syncthreads();

    Pack2 p[8];
#pragma unroll
    for (int i = 0; i < 8; ++i) {
        p[i].h[0] = t[c + 2 * i][r];
        p[i].h[1] = t[c + 2 * i + 1][r];
    }
    __bf16* dst = Wt + (size_t)(n0 + r) * DIM + k0 + c;
    *(uint4*)dst       = make_uint4(p[0].u, p[1].u, p[2].u, p[3].u);
    *(uint4*)(dst + 8) = make_uint4(p[4].u, p[5].u, p[6].u, p[7].u);
}

// A2[b,h,n*32+d] = P[b,n,h*32+d] * V[b,h,n*32+d], emitted directly as bf16
__global__ __launch_bounds__(256)
void ew_combine(const float* __restrict__ P, const float* __restrict__ V,
                __bf16* __restrict__ A2)
{
    const int bh = blockIdx.x;           // 0..6271
    const int b  = bh / 196;
    const int h  = bh % 196;
    const float* Pb = P + (size_t)b * 196 * DIM;
    const float* Vr = V + (size_t)bh * DIM;
    __bf16*      Ar = A2 + (size_t)bh * DIM;
    for (int j = threadIdx.x * 4; j < DIM; j += 256 * 4) {
        const int n = j >> 5;
        const int d = j & 31;
        float4 p = *(const float4*)(Pb + (size_t)n * DIM + h * 32 + d);
        float4 v = *(const float4*)(Vr + j);
        Pack2 lo, hi;
        lo.h[0] = (__bf16)(p.x * v.x); lo.h[1] = (__bf16)(p.y * v.y);
        hi.h[0] = (__bf16)(p.z * v.z); hi.h[1] = (__bf16)(p.w * v.w);
        *(uint2*)(Ar + j) = make_uint2(lo.u, hi.u);
    }
}

extern "C" void kernel_launch(void* const* d_in, const int* in_sizes, int n_in,
                              void* d_out, int out_size, void* d_ws, size_t ws_size,
                              hipStream_t stream)
{
    const float* x  = (const float*)d_in[0];
    const float* wq = (const float*)d_in[1];
    const float* wk = (const float*)d_in[2];
    const float* wv = (const float*)d_in[3];
    const float* wo = (const float*)d_in[4];
    const float* bo = (const float*)d_in[5];
    float* out = (float*)d_out;

    const size_t elems = (size_t)32 * 196 * DIM;   // 39,337,984 per matrix
    float*  P   = (float*)d_ws;                    // relu(q)*relu(k), fp32
    float*  V   = P + elems;                       // v projection, fp32
    __bf16* xb  = (__bf16*)(V + elems);            // X in bf16
    __bf16* wtb = xb + elems;                      // current weight, bf16 transposed
    __bf16* a2b = wtb + elems;                     // A2 in bf16

    dim3 gG(DIM / BM, DIM / BN);                   // 49 x 49
    dim3 gT(DIM / 64, DIM / 64);                   // 98 x 98
    dim3 blk(256);
    const int nConvBlk = (int)(elems / 1024);      // 38416

    convert_bf16<<<nConvBlk, blk, 0, stream>>>(x, xb);

    convert_tr_bf16<<<gT, blk, 0, stream>>>(wq, wtb);
    gemm_bf16_wmma<1><<<gG, blk, 0, stream>>>(xb, wtb, P, nullptr, nullptr);  // P = relu(X Wq)

    convert_tr_bf16<<<gT, blk, 0, stream>>>(wk, wtb);
    gemm_bf16_wmma<2><<<gG, blk, 0, stream>>>(xb, wtb, P, P, nullptr);        // P *= relu(X Wk)

    convert_tr_bf16<<<gT, blk, 0, stream>>>(wv, wtb);
    gemm_bf16_wmma<0><<<gG, blk, 0, stream>>>(xb, wtb, V, nullptr, nullptr);  // V = X Wv

    ew_combine<<<dim3(32 * 196), blk, 0, stream>>>(P, V, a2b);                // A2 (bf16)

    convert_tr_bf16<<<gT, blk, 0, stream>>>(wo, wtb);
    gemm_bf16_wmma<3><<<gG, blk, 0, stream>>>(a2b, wtb, out, nullptr, bo);    // Out = A2 Wo + b
}
